// SpatialCorrelationSampler_27814208209674
// MI455X (gfx1250) — matrix operand order repaired and checked by reading, not compiled, not run on hardware
//
#include <hip/hip_runtime.h>

// CDNA5 / gfx1250 correlation (cost-volume) kernel using f16 WMMA with f32 accumulate.
// out[b,ph,pw,h,w] = sum_c in1[b,c,h,w] * in2[b,c,h+ph-4,w+pw-4], zero-padded.

typedef _Float16 v16h __attribute__((ext_vector_type(16)));
typedef _Float16 h8   __attribute__((ext_vector_type(8)));
typedef float    v8f  __attribute__((ext_vector_type(8)));

#define BATCH  16
#define CCH    256
#define HH     96
#define WW     96
#define TH     8          // output rows per workgroup (== waves per block)
#define TW     16         // output pixels per workgroup (one WMMA M tile)
#define KC     32         // channels per chunk (one WMMA K step)
#define NCHUNK (CCH / KC) // 8
#define P2ROWS (TH + 8)   // 16 in2 rows staged (with +/-4 halo)
#define P2PX   (TW + 8)   // 24 in2 pixels staged (with +/-4 halo)
#define LSTR   40         // halfs per pixel row in LDS (32 data + 8 pad -> 80B, conflict-free b128)

__global__ __launch_bounds__(256)
void corr_wmma_kernel(const float* __restrict__ in1,
                      const float* __restrict__ in2,
                      float* __restrict__ out)
{
    // LDS staging, [row][pixel][channel] layout, pixel stride 80B
    __shared__ alignas(16) _Float16 lds1[TH * TW * LSTR];        // in1: 10.0 KB
    __shared__ alignas(16) _Float16 lds2[P2ROWS * P2PX * LSTR];  // in2: 30.0 KB

    const int tid  = threadIdx.x;
    const int lane = tid & 31;
    const int iw   = tid >> 5;        // wave id == output row within tile
    const int ln16 = lane & 15;
    const int sel  = lane >> 4;       // 0: lanes 0-15, 1: lanes 16-31

    const int w0 = blockIdx.x * TW;
    const int h0 = blockIdx.y * TH;
    const int b  = blockIdx.z;

    const v8f vzero = {0.f, 0.f, 0.f, 0.f, 0.f, 0.f, 0.f, 0.f};
    v8f acc[9][2];                    // 9 row-shifts x 2 band tiles, f32 16x16 each
#pragma unroll
    for (int r = 0; r < 9; ++r)
#pragma unroll
        for (int t = 0; t < 2; ++t)
            acc[r][t] = vzero;

    for (int ck = 0; ck < NCHUNK; ++ck) {
        const int c0 = ck * KC;

        // ---- stage in1: TH x TW x KC (always in-bounds), fp32 -> f16 ----
#pragma unroll
        for (int k = 0; k < 2; ++k) {
            const int u   = tid + k * 256;
            const int px  = u & 15;
            const int tmp = u >> 4;
            const int oct = tmp & 3;          // which group of 8 channels
            const int row = tmp >> 2;
            const float* p = in1 + (((size_t)b * CCH + (c0 + oct * 8)) * HH + (h0 + row)) * WW
                                 + (w0 + px);
            h8 hv;
#pragma unroll
            for (int l = 0; l < 8; ++l)
                hv[l] = (_Float16)p[l * (HH * WW)];
            *(h8*)&lds1[(row * TW + px) * LSTR + oct * 8] = hv;
        }

        // ---- stage in2: P2ROWS x P2PX x KC with zero-padded halo ----
#pragma unroll
        for (int k = 0; k < 6; ++k) {
            const int u   = tid + k * 256;
            const int px  = u % 24;
            const int tmp = u / 24;
            const int oct = tmp & 3;
            const int row = tmp >> 2;
            const int hg  = h0 + row - 4;
            const int wg  = w0 + px - 4;
            const bool inb = (hg >= 0) & (hg < HH) & (wg >= 0) & (wg < WW);
            const int hc = inb ? hg : 0;      // clamped, always-valid address
            const int wc = inb ? wg : 0;
            const float zmask = inb ? 1.0f : 0.0f;
            const float* p = in2 + (((size_t)b * CCH + (c0 + oct * 8)) * HH + hc) * WW + wc;
            h8 hv;
#pragma unroll
            for (int l = 0; l < 8; ++l)
                hv[l] = (_Float16)(p[l * (HH * WW)] * zmask);   // branch-free zero pad
            *(h8*)&lds2[(row * P2PX + px) * LSTR + oct * 8] = hv;
            // prefetch next channel chunk (global_prefetch_b8)
            if (ck + 1 < NCHUNK)
                __builtin_prefetch(p + KC * (HH * WW), 0, 3);
        }
        __syncthreads();

        // ---- A fragment: in1 row iw, 16 px x 32 ch (ISA 16-bit A 16x32 layout) ----
        // lane<16: M=lane, halfs {0..7},{16..23}; lane>=16: M=lane-16, halfs {8..15},{24..31}
        const _Float16* abase = &lds1[(iw * TW + ln16) * LSTR];
        h8 alo = *(const h8*)(abase + sel * 8);
        h8 ahi = *(const h8*)(abase + 16 + sel * 8);
        v16h A = __builtin_shufflevector(alo, ahi,
                                         0, 1, 2, 3, 4, 5, 6, 7, 8, 9, 10, 11, 12, 13, 14, 15);

        // ---- 18 WMMAs (9 row-shifts x 2 band tiles), depth-1 software pipeline on B ----
        // B 16-bit 32x16 layout: lane<16: N=lane, halfs {0..7},{16..23}; lane>=16: halfs {8..15},{24..31}
        h8 bl0, bh0, bl1, bh1;
        {
            const _Float16* bb = &lds2[((iw + 0) * P2PX + (0 * 8 + ln16)) * LSTR + sel * 8];
            bl0 = *(const h8*)(bb);
            bh0 = *(const h8*)(bb + 16);
        }
#pragma unroll
        for (int idx = 0; idx < 18; ++idx) {
            if (idx + 1 < 18) {               // issue next B-fragment loads before consuming
                const int rn = (idx + 1) >> 1;
                const int tn = (idx + 1) & 1;
                const _Float16* bb =
                    &lds2[((iw + rn) * P2PX + (tn * 8 + ln16)) * LSTR + sel * 8];
                bl1 = *(const h8*)(bb);
                bh1 = *(const h8*)(bb + 16);
            }
            v16h Bf = __builtin_shufflevector(bl0, bh0,
                                              0, 1, 2, 3, 4, 5, 6, 7, 8, 9, 10, 11, 12, 13, 14, 15);
            acc[idx >> 1][idx & 1] = __builtin_amdgcn_wmma_f32_16x16x32_f16(
                false, A, false, Bf, (short)0, acc[idx >> 1][idx & 1], false, false);
            bl0 = bl1;
            bh0 = bh1;
        }
        __syncthreads();
    }

    // ---- scatter results: D[i][j]; lane holds (i = v + 8*sel, j = ln16) per C/D layout ----
    const int h = h0 + iw;
#pragma unroll
    for (int r = 0; r < 9; ++r) {
#pragma unroll
        for (int t = 0; t < 2; ++t) {
#pragma unroll
            for (int v = 0; v < 8; ++v) {
                const int i    = v + sel * 8;
                const int j    = ln16;
                const int diff = j - i;
                const int dw   = diff + (t ? 4 : -4);
                // disjoint ownership of the 9-wide band between the two tiles
                const bool emit = t ? ((j >= 8) & (diff >= -8) & (diff <= 0))
                                    : ((diff >= 0) & (diff <= 8));
                if (emit) {
                    const int pw = dw + 4;
                    const size_t idx =
                        ((((size_t)b * 9 + r) * 9 + pw) * HH + h) * WW + (w0 + i);
                    out[idx] = acc[r][t][v];
                }
            }
        }
    }
}

extern "C" void kernel_launch(void* const* d_in, const int* in_sizes, int n_in,
                              void* d_out, int out_size, void* d_ws, size_t ws_size,
                              hipStream_t stream) {
    const float* in1 = (const float*)d_in[0];
    const float* in2 = (const float*)d_in[1];
    float* outp = (float*)d_out;
    dim3 grid(WW / TW, HH / TH, BATCH);   // (6, 12, 16)
    corr_wmma_kernel<<<grid, 256, 0, stream>>>(in1, in2, outp);
}
